// GIN_29661044146327
// MI455X (gfx1250) — matrix operand order repaired and checked by reading, not compiled, not run on hardware
//
#include <hip/hip_runtime.h>

#define NODES 100000
#define FEATS 512
#define NEDGES 800000

typedef __attribute__((ext_vector_type(16))) _Float16 v16h;
typedef __attribute__((ext_vector_type(8)))  float    v8f;
typedef __attribute__((ext_vector_type(4)))  unsigned u32x4;

// GEMM tile config
#define BM 128
#define BN 128
#define BK 32
#define APITCH 40    // f16 elems per A-LDS row (32 + 8 pad) -> 80B, 16B aligned
#define BPITCH 520   // f16 elems per B-LDS row (512 + 8 pad) -> 1040B, 16B aligned

// A fragment (16x32 f16): lane l holds m = l%16, khalf = l/16.
// elems 0..7  -> k = khalf*16 + 0..7   (contiguous)
// elems 8..15 -> k = khalf*16 + 8..15  (contiguous)
__device__ inline v16h ld_frag_a(const _Float16* row, int khalf) {
    v16h r;
    ((u32x4*)&r)[0] = *(const u32x4*)(row + khalf * 16);
    ((u32x4*)&r)[1] = *(const u32x4*)(row + khalf * 16 + 8);
    return r;
}

// B fragment (32x16 f16): lane l holds n = l%16, k = (l/16)*16 + 0..15 contiguous.
// Bs row is [n][k] so this is one 32B contiguous run.
__device__ inline v16h ld_frag_b(const _Float16* row, int koff) {
    v16h r;
    ((u32x4*)&r)[0] = *(const u32x4*)(row + koff);
    ((u32x4*)&r)[1] = *(const u32x4*)(row + koff + 8);
    return r;
}

template <bool AF32, bool RELU, bool OUTF16>
__global__ __launch_bounds__(256) void gemm_wmma(
    const void* __restrict__ Aglob,   // [M][512] f32 (AF32) or f16
    const _Float16* __restrict__ Bt,  // [512][512] f16, Bt[n][k] = W[k][n]
    const float* __restrict__ bias,   // [512]
    void* __restrict__ Out,           // [M][512] f32 or f16
    int M)
{
    __shared__ _Float16 Bs[BN * BPITCH];  // 133,120 B: full-K weight slice, resident
    __shared__ _Float16 As[BM * APITCH];  // 10,240 B: per-step A tile

    const int tid   = threadIdx.x;
    const int lane  = tid & 31;
    const int w     = tid >> 5;
    const int wm    = w & 3;    // 32-row group
    const int wn    = w >> 2;   // 64-col group
    const int mBase = blockIdx.y * BM;
    const int nBase = blockIdx.x * BN;

    // ---- preload full-K B slice (128 cols x 512 k) into LDS, 16B chunks ----
    for (int c = tid; c < (BN * FEATS) / 8; c += 256) {
        int n  = c >> 6;          // 64 chunks of 8 f16 per row
        int kc = (c & 63) << 3;
        u32x4 v = *(const u32x4*)(Bt + (size_t)(nBase + n) * FEATS + kc);
        *(u32x4*)(Bs + n * BPITCH + kc) = v;
    }
    __syncthreads();

    v8f acc[2][4] = {};

    const int mLoad = tid >> 1;        // row 0..127 staged by this thread
    const int kLoad = (tid & 1) << 4;  // k chunk 0 or 16

    for (int k0 = 0; k0 < FEATS; k0 += BK) {
        // ---- stage A tile [BM x BK] into LDS as f16 ----
        {
            const int row = mBase + mLoad;
            _Float16* dstp = As + mLoad * APITCH + kLoad;
            if (AF32) {
                const float* ap = (const float*)Aglob + (size_t)row * FEATS + k0 + kLoad;
                if (row < M) {
#pragma unroll
                    for (int i = 0; i < 16; i += 4) {
                        float4 v = *(const float4*)(ap + i);
                        dstp[i + 0] = (_Float16)v.x; dstp[i + 1] = (_Float16)v.y;
                        dstp[i + 2] = (_Float16)v.z; dstp[i + 3] = (_Float16)v.w;
                    }
                } else {
#pragma unroll
                    for (int i = 0; i < 16; ++i) dstp[i] = (_Float16)0.f;
                }
            } else {
                const _Float16* ap = (const _Float16*)Aglob + (size_t)row * FEATS + k0 + kLoad;
                if (row < M) {
                    ((u32x4*)dstp)[0] = ((const u32x4*)ap)[0];
                    ((u32x4*)dstp)[1] = ((const u32x4*)ap)[1];
                } else {
                    u32x4 zed = {0u, 0u, 0u, 0u};
                    ((u32x4*)dstp)[0] = zed;
                    ((u32x4*)dstp)[1] = zed;
                }
            }
        }
        __syncthreads();

        // prefetch next A tile into cache (global_prefetch_b8)
        if (k0 + BK < FEATS) {
            const char* pf = AF32
                ? (const char*)((const float*)Aglob + (size_t)(mBase + mLoad) * FEATS + k0 + BK)
                : (const char*)((const _Float16*)Aglob + (size_t)(mBase + mLoad) * FEATS + k0 + BK);
            __builtin_prefetch(pf, 0, 0);
        }

        const int khalf = lane >> 4;
        const int lr    = lane & 15;
        v16h afrag[2], bfrag[4];
#pragma unroll
        for (int mt = 0; mt < 2; ++mt)
            afrag[mt] = ld_frag_a(As + (wm * 32 + mt * 16 + lr) * APITCH, khalf);
#pragma unroll
        for (int nt = 0; nt < 4; ++nt)
            bfrag[nt] = ld_frag_b(Bs + (wn * 64 + nt * 16 + lr) * BPITCH + k0, khalf * 16);

#pragma unroll
        for (int mt = 0; mt < 2; ++mt)
#pragma unroll
            for (int nt = 0; nt < 4; ++nt)
                acc[mt][nt] = __builtin_amdgcn_wmma_f32_16x16x32_f16(
                    false, afrag[mt], false, bfrag[nt],
                    (short)0, acc[mt][nt], false, false);

        __syncthreads();
    }

    // ---- epilogue: C/D layout: VGPR v, lanes 0-15 -> M=v, lanes 16-31 -> M=v+8; N=lane%16
    const int lr = lane & 15;
    const int lh = lane >> 4;
#pragma unroll
    for (int mt = 0; mt < 2; ++mt) {
#pragma unroll
        for (int nt = 0; nt < 4; ++nt) {
            const int col = nBase + wn * 64 + nt * 16 + lr;
            const float b = bias[col];
#pragma unroll
            for (int v = 0; v < 8; ++v) {
                const int row = mBase + wm * 32 + mt * 16 + lh * 8 + v;
                if (row < M) {
                    float val = acc[mt][nt][v] + b;
                    if (RELU) val = val > 0.f ? val : 0.f;
                    if (OUTF16)
                        ((_Float16*)Out)[(size_t)row * FEATS + col] = (_Float16)val;
                    else
                        ((float*)Out)[(size_t)row * FEATS + col] = val;
                }
            }
        }
    }
}

// z = h (vectorized copy)
__global__ void copy_f32(const float* __restrict__ in, float* __restrict__ out, size_t n) {
    size_t i = ((size_t)blockIdx.x * blockDim.x + threadIdx.x) * 4;
    if (i < n) *(float4*)(out + i) = *(const float4*)(in + i);
}

// z[dst] += h[src] : 128 threads per edge, float4 gather + 4 f32 atomics
__global__ void scatter_add(const float* __restrict__ h, const int* __restrict__ src,
                            const int* __restrict__ dst, float* __restrict__ z, int nE) {
    long long t = (long long)blockIdx.x * blockDim.x + threadIdx.x;
    int e = (int)(t >> 7);
    if (e >= nE) return;
    int c = ((int)t & 127) << 2;
    int s = src[e], d = dst[e];
    float4 v = *(const float4*)(h + (size_t)s * FEATS + c);
    float* zp = z + (size_t)d * FEATS + c;
    atomicAdd(zp + 0, v.x);
    atomicAdd(zp + 1, v.y);
    atomicAdd(zp + 2, v.z);
    atomicAdd(zp + 3, v.w);
}

// Wt[n][k] = (f16) W[k][n]  (512x512)
__global__ void wt_transpose(const float* __restrict__ W, _Float16* __restrict__ Wt) {
    int t = blockIdx.x * blockDim.x + threadIdx.x;
    int n = t >> 9;
    int k = t & 511;
    Wt[(size_t)n * FEATS + k] = (_Float16)W[(size_t)k * FEATS + n];
}

extern "C" void kernel_launch(void* const* d_in, const int* in_sizes, int n_in,
                              void* d_out, int out_size, void* d_ws, size_t ws_size,
                              hipStream_t stream) {
    const float* feats = (const float*)d_in[0];
    const int* src = (const int*)d_in[1];
    const int* dst = (const int*)d_in[2];
    const float *W1[3], *b1[3], *W2[3], *b2[3];
    for (int i = 0; i < 3; ++i) {
        W1[i] = (const float*)d_in[3 + i * 4 + 0];
        b1[i] = (const float*)d_in[3 + i * 4 + 1];
        W2[i] = (const float*)d_in[3 + i * 4 + 2];
        b2[i] = (const float*)d_in[3 + i * 4 + 3];
    }

    char* ws = (char*)d_ws;
    const size_t nodeF32 = (size_t)NODES * FEATS * 4;  // 204.8 MB
    const size_t nodeF16 = (size_t)NODES * FEATS * 2;  // 102.4 MB
    const size_t wtF16   = (size_t)FEATS * FEATS * 2;

    float* z = (float*)ws;        ws += nodeF32;
    _Float16* y = (_Float16*)ws;  ws += nodeF16;
    float* hA = (float*)ws;       ws += nodeF32;
    float* hB = (float*)ws;       ws += nodeF32;
    _Float16* Wt[6];
    for (int i = 0; i < 6; ++i) { Wt[i] = (_Float16*)ws; ws += wtF16; }

    // one-time (per launch) weight convert+transpose to f16
    for (int i = 0; i < 3; ++i) {
        wt_transpose<<<1024, 256, 0, stream>>>(W1[i], Wt[2 * i + 0]);
        wt_transpose<<<1024, 256, 0, stream>>>(W2[i], Wt[2 * i + 1]);
    }

    dim3 gGemm(FEATS / BN, (NODES + BM - 1) / BM);  // (4, 782)
    const size_t nElems = (size_t)NODES * FEATS;

    const float* hin = feats;
    float* houts[3] = { hA, hB, (float*)d_out };
    for (int layer = 0; layer < 3; ++layer) {
        copy_f32<<<(unsigned)(nElems / 4 / 256), 256, 0, stream>>>(hin, z, nElems);
        scatter_add<<<(NEDGES * 128) / 256, 256, 0, stream>>>(hin, src, dst, z, NEDGES);
        // GEMM1: y = relu(z @ W1 + b1), f32 A -> f16 out
        gemm_wmma<true, true, true><<<gGemm, 256, 0, stream>>>(
            z, Wt[2 * layer], b1[layer], y, NODES);
        // GEMM2: h_next = y @ W2 + b2 (+relu except final), f16 A -> f32 out
        if (layer < 2)
            gemm_wmma<false, true, false><<<gGemm, 256, 0, stream>>>(
                y, Wt[2 * layer + 1], b2[layer], houts[layer], NODES);
        else
            gemm_wmma<false, false, false><<<gGemm, 256, 0, stream>>>(
                y, Wt[2 * layer + 1], b2[layer], houts[layer], NODES);
        hin = houts[layer];
    }
}
